// SkipBCell_61890478735975
// MI455X (gfx1250) — matrix operand order repaired and checked by reading, not compiled, not run on hardware
//
#include <hip/hip_runtime.h>
#include <cstdint>
#include <cstddef>

typedef __attribute__((ext_vector_type(16))) __bf16 v16bf;
typedef __attribute__((ext_vector_type(8)))  float  v8f;
typedef __attribute__((ext_vector_type(4)))  int    v4i_t;

#define T_  512
#define Bb  64
#define Ii  1024
#define Hh  1024
#define G4H 4096
#define KC  32      // K-depth per WMMA chunk
#define NT  64      // h-columns per block

__device__ __forceinline__ float sigm(float x) { return 1.0f / (1.0f + __expf(-x)); }

// ---------------------------------------------------------------------------
// CDNA5 async global->LDS copy (GLOBAL_LOAD_ASYNC_TO_LDS_B128, ASYNCcnt).
// Builtin signature (from clang diagnostic): (v4i AS1*, v4i AS3*, imm, imm).
// Generic->AS3 cast: low 32 bits of a generic LDS address are the LDS byte
// offset (ISA 10.2 aperture rules), AS3 pointers are 32-bit.
// ---------------------------------------------------------------------------
#if defined(__has_builtin)
#  if __has_builtin(__builtin_amdgcn_global_load_async_to_lds_b128)
#    define HAVE_ASYNC_LDS 1
#  endif
#endif
#ifndef HAVE_ASYNC_LDS
#  define HAVE_ASYNC_LDS 0
#endif

typedef __attribute__((address_space(1))) v4i_t as1_v4i;
typedef __attribute__((address_space(3))) v4i_t as3_v4i;

__device__ __forceinline__ void copy16_async(const void* g, void* l) {
#if HAVE_ASYNC_LDS
    __builtin_amdgcn_global_load_async_to_lds_b128(
        (as1_v4i*)(uintptr_t)g,
        (as3_v4i*)(uint32_t)(uintptr_t)l,
        0, 0);
#else
    *reinterpret_cast<uint4*>(l) = *reinterpret_cast<const uint4*>(g);
#endif
}

__device__ __forceinline__ void async_wait0() {
#if HAVE_ASYNC_LDS
#  if __has_builtin(__builtin_amdgcn_s_wait_asynccnt)
    __builtin_amdgcn_s_wait_asynccnt(0);
#  else
    asm volatile("s_wait_asynccnt 0" ::: "memory");
#  endif
#endif
}

// ---------------------------------------------------------------------------
// One-time setup: fp32 -> bf16 conversion of input + weights, fused bias,
// state init.
// ---------------------------------------------------------------------------
__global__ __launch_bounds__(256) void init_ws_kernel(
    const float* __restrict__ input,
    const float* __restrict__ Wih, const float* __restrict__ Whh, const float* __restrict__ Wsk,
    const float* __restrict__ bih, const float* __restrict__ bhh,
    const float* __restrict__ hidden, const float* __restrict__ candidate,
    __bf16* __restrict__ x_bf,
    __bf16* __restrict__ Wih_bf, __bf16* __restrict__ Whh_bf, __bf16* __restrict__ Wsk_bf,
    float* __restrict__ bias, __bf16* __restrict__ h0, __bf16* __restrict__ sh0,
    float* __restrict__ c0)
{
    size_t i0 = (size_t)blockIdx.x * blockDim.x + threadIdx.x;
    size_t stride = (size_t)gridDim.x * blockDim.x;
    const size_t NX = (size_t)T_ * Bb * Ii;
    for (size_t i = i0; i < NX; i += stride) x_bf[i] = (__bf16)input[i];
    const size_t NW = (size_t)G4H * Ii;
    for (size_t i = i0; i < NW; i += stride) {
        Wih_bf[i] = (__bf16)Wih[i];
        Whh_bf[i] = (__bf16)Whh[i];
        Wsk_bf[i] = (__bf16)Wsk[i];
    }
    for (size_t i = i0; i < (size_t)G4H; i += stride) bias[i] = bih[i] + bhh[i];
    for (size_t i = i0; i < (size_t)Bb * Hh; i += stride) {
        __bf16 hb = (__bf16)hidden[i];
        h0[i] = hb;
        sh0[i] = hb;
        c0[i] = candidate[i];
    }
}

// ---------------------------------------------------------------------------
// One LSTM timestep. Grid: 16 blocks (each owns 64 h-columns). 256 threads =
// 8 wave32 waves; wave w handles gate g = w>>1, column-half ch = w&1.
// Double-buffered LDS staging via async global->LDS, overlapped with
// v_wmma_f32_16x16x32_bf16 compute on the other buffer.
// ---------------------------------------------------------------------------
__global__ __launch_bounds__(256) void lstm_step_kernel(
    const __bf16* __restrict__ x_bf,    // [B, I] bf16, this timestep's input
    const __bf16* __restrict__ Wih,     // [4H, I] bf16
    const __bf16* __restrict__ Whh,     // [4H, H] bf16
    const __bf16* __restrict__ Wsk,     // [4H, H] bf16
    const float*  __restrict__ bias,    // [4H] fp32 (bih + bhh)
    const __bf16* __restrict__ h_rd,    // [B, H] bf16, h_{t-1}
    __bf16*       __restrict__ h_wr,    // [B, H] bf16, h_t (ping-pong)
    const __bf16* __restrict__ sh_rd,   // [B, H] bf16, skip-hidden state
    __bf16*       __restrict__ sh_wr,   // [B, H] bf16 (written only if fire)
    float*        __restrict__ c_state, // [B, H] fp32, in/out
    float*        __restrict__ out_t,   // [B, H] fp32 -> outputs[t]
    float*        __restrict__ out_h,   // final h (t == T-1) or nullptr
    float*        __restrict__ out_c,   // final c (t == T-1) or nullptr
    int fire)
{
    // 64 KB LDS, two overlapping phases:
    //   GEMM phase:    As[2][64][32] bf16 (8 KB) + Bs[2][256][32] bf16 (32 KB)
    //   combine phase: Gs [4][64][64] f32 (64 KB)
    __shared__ __align__(16) char smem[65536];
    __bf16* As = reinterpret_cast<__bf16*>(smem);           // + buf*2048 elems
    __bf16* Bs = reinterpret_cast<__bf16*>(smem + 8192);    // + buf*8192 elems
    float*  Gs = reinterpret_cast<float*>(smem);

    const int tid  = threadIdx.x;
    const int lane = tid & 31;
    const int wv   = tid >> 5;
    const int g    = wv >> 1;   // gate 0..3
    const int ch   = wv & 1;    // 32-column half
    const int col0 = blockIdx.x * NT;

    const __bf16* Asrc[3] = { x_bf, h_rd, sh_rd };
    const __bf16* Bsrc[3] = { Wih, Whh, Wsk };

    // Accumulators: 4 row-tiles (M=64) x 2 col-tiles (16 each) of 16x16 f32.
    // C/D layout: vgpr r of lane l holds (m = r + 8*(l>=16), n = l&15).
    v8f acc[4][2];
#pragma unroll
    for (int j = 0; j < 2; ++j) {
        float bv = bias[g * Hh + col0 + ch * 32 + j * 16 + (lane & 15)];
#pragma unroll
        for (int m = 0; m < 4; ++m)
#pragma unroll
            for (int r = 0; r < 8; ++r) acc[m][j][r] = bv;
    }

    // Per-thread staging assignments (constant across chunks).
    const int arow = tid >> 2, aseg = tid & 3;   // A: 64 rows x 4 x 16B

    const int NC = fire ? 96 : 64;               // flattened (segment, kchunk)

    // ---- stage(chunk ci) into buffer b: issue async copies, no wait ----
    auto stage = [&](int ci, int b) {
        const int s  = ci >> 5;
        const int kb = (ci & 31) * KC;
        // A tile: 64 x 32 bf16
        copy16_async(Asrc[s] + (size_t)arow * 1024 + kb + aseg * 8,
                     As + b * 2048 + arow * KC + aseg * 8);
        // B tile: 256 gate-rows x 32 bf16
#pragma unroll
        for (int i = 0; i < 4; ++i) {
            int lin  = tid + 256 * i;            // 0..1023
            int brow = lin >> 2, seg = lin & 3;
            int gg = brow >> 6, rr = brow & 63;
            copy16_async(Bsrc[s] + (size_t)(gg * Hh + col0 + rr) * 1024 + kb + seg * 8,
                         Bs + b * 8192 + brow * KC + seg * 8);
        }
    };

    stage(0, 0);
    async_wait0();
    __syncthreads();

    for (int ci = 0; ci < NC; ++ci) {
        const int buf = ci & 1;
        if (ci + 1 < NC) stage(ci + 1, buf ^ 1);   // overlap with compute

        const __bf16* Ab = As + buf * 2048;
        const __bf16* Bbuf = Bs + buf * 8192;

        // A fragments (16-bit A 16x32 layout: lanes 0-15 carry K 0..7,16..23
        // of row M=l; lanes 16-31 carry K 8..15,24..31)
        v16bf afrag[4];
        const int ak = (lane < 16) ? 0 : 8;
#pragma unroll
        for (int m = 0; m < 4; ++m) {
            const __bf16* ap = Ab + (m * 16 + (lane & 15)) * KC;
            union { v16bf v; uint4 q[2]; } u;
            u.q[0] = *reinterpret_cast<const uint4*>(ap + ak);
            u.q[1] = *reinterpret_cast<const uint4*>(ap + ak + 16);
            afrag[m] = u.v;
        }

        // B fragments (32x16 B: lanes 0-15 K=0..7,16..23 of col l&15, lanes 16-31
        // K=8..15,24..31) + 8 WMMAs
        const int bk = (lane < 16) ? 0 : 8;
#pragma unroll
        for (int j = 0; j < 2; ++j) {
            int brow = g * 64 + ch * 32 + j * 16 + (lane & 15);
            const __bf16* bp = Bbuf + brow * KC + bk;
            union { v16bf v; uint4 q[2]; } u;
            u.q[0] = *reinterpret_cast<const uint4*>(bp);
            u.q[1] = *reinterpret_cast<const uint4*>(bp + 16);
            v16bf bfrag = u.v;
#pragma unroll
            for (int m = 0; m < 4; ++m)
                acc[m][j] = __builtin_amdgcn_wmma_f32_16x16x32_bf16(
                    false, afrag[m], false, bfrag,
                    (short)0, acc[m][j], false, false);
        }

        async_wait0();        // next chunk's copies have landed in LDS
        __syncthreads();      // safe to read buf^1 / overwrite buf next iter
    }

    // ---- exchange gates across waves through LDS (overlays staging) ----
#pragma unroll
    for (int m = 0; m < 4; ++m)
#pragma unroll
        for (int j = 0; j < 2; ++j)
#pragma unroll
            for (int r = 0; r < 8; ++r) {
                int mrow = m * 16 + r + ((lane >> 4) << 3);
                int ncol = ch * 32 + j * 16 + (lane & 15);
                Gs[(g * 64 + mrow) * 64 + ncol] = acc[m][j][r];
            }
    __syncthreads();

    // ---- elementwise LSTM cell: 64x64 cells, 16 per thread ----
#pragma unroll 4
    for (int it = 0; it < 16; ++it) {
        int idx = tid + it * 256;
        int b = idx >> 6, n = idx & 63;
        int nc = col0 + n;
        float ig = Gs[(0 * 64 + b) * 64 + n];
        float fg = Gs[(1 * 64 + b) * 64 + n];
        float cs = Gs[(2 * 64 + b) * 64 + n];
        float og = Gs[(3 * 64 + b) * 64 + n];
        size_t off = (size_t)b * Hh + nc;
        float c_old = c_state[off];
        float cn = sigm(fg) * c_old + sigm(ig) * tanhf(cs);
        float h  = sigm(og) * tanhf(cn);
        c_state[off] = cn;
        out_t[off]   = h;
        __bf16 hb = (__bf16)h;
        h_wr[off] = hb;
        if (fire) sh_wr[off] = hb;
        if (out_h) { out_h[off] = h; out_c[off] = cn; }
    }
}

// ---------------------------------------------------------------------------
// Host launcher: init + 512 stream-ordered step kernels (graph-capturable).
// ---------------------------------------------------------------------------
extern "C" void kernel_launch(void* const* d_in, const int* in_sizes, int n_in,
                              void* d_out, int out_size, void* d_ws, size_t ws_size,
                              hipStream_t stream) {
    (void)in_sizes; (void)n_in; (void)out_size; (void)ws_size;
    const float* input     = (const float*)d_in[0];  // [T, B, I]
    const float* hidden    = (const float*)d_in[1];  // [B, H]
    const float* candidate = (const float*)d_in[2];  // [B, H]
    const float* Wih       = (const float*)d_in[3];  // [4H, I]
    const float* Whh       = (const float*)d_in[4];  // [4H, H]
    const float* b_ih      = (const float*)d_in[5];  // [4H]
    const float* b_hh      = (const float*)d_in[6];  // [4H]
    const float* Wsk       = (const float*)d_in[7];  // [4H, H]
    float* out = (float*)d_out;

    // workspace carve-up (256B aligned)
    char* ws = (char*)d_ws;
    size_t off = 0;
    auto carve = [&](size_t bytes) {
        void* p = ws + off;
        off = (off + bytes + 255) & ~(size_t)255;
        return p;
    };
    const size_t NX  = (size_t)T_ * Bb * Ii;  // input elements
    const size_t WB  = (size_t)G4H * Ii;      // weight elements
    const size_t SH  = (size_t)Bb * Hh;       // state elements
    __bf16* x_bf   = (__bf16*)carve(NX * 2);
    __bf16* Wih_bf = (__bf16*)carve(WB * 2);
    __bf16* Whh_bf = (__bf16*)carve(WB * 2);
    __bf16* Wsk_bf = (__bf16*)carve(WB * 2);
    float*  bias   = (float*)carve((size_t)G4H * 4);
    __bf16* h_bf   = (__bf16*)carve(2 * SH * 2);   // ping-pong
    __bf16* sh_bf  = (__bf16*)carve(2 * SH * 2);   // ping-pong
    float*  c_st   = (float*)carve(SH * 4);

    init_ws_kernel<<<4096, 256, 0, stream>>>(
        input, Wih, Whh, Wsk, b_ih, b_hh, hidden, candidate,
        x_bf, Wih_bf, Whh_bf, Wsk_bf, bias, h_bf, sh_bf, c_st);

    for (int t = 0; t < T_; ++t) {
        int fire = ((t % 5) == 4) ? 1 : 0;
        int hr = t & 1, hw = hr ^ 1;
        int sr = (t / 5) & 1, sw = sr ^ 1;
        float* fh = (t == T_ - 1) ? (out + (size_t)T_ * SH) : nullptr;
        float* fc = (t == T_ - 1) ? (out + (size_t)T_ * SH + SH) : nullptr;
        lstm_step_kernel<<<Hh / NT, 256, 0, stream>>>(
            x_bf + (size_t)t * Bb * Ii,
            Wih_bf, Whh_bf, Wsk_bf, bias,
            h_bf + (size_t)hr * SH, h_bf + (size_t)hw * SH,
            sh_bf + (size_t)sr * SH, sh_bf + (size_t)sw * SH,
            c_st,
            out + (size_t)t * SH,
            fh, fc, fire);
    }
}
